// DiscriptorMatchLoss_19344532702283
// MI455X (gfx1250) — matrix operand order repaired and checked by reading, not compiled, not run on hardware
//
#include <hip/hip_runtime.h>
#include <math.h>

typedef __attribute__((ext_vector_type(16))) _Float16 v16h;
typedef __attribute__((ext_vector_type(8)))  _Float16 v8h;
typedef __attribute__((ext_vector_type(8)))  float    v8f;

#define WS_HDR 256  // bytes reserved at front of d_ws for {sum, count}

__global__ void dml_init_accum(float* acc) {
    if (threadIdx.x < 2) acc[threadIdx.x] = 0.0f;
}

// One wave per descriptor row: L2-normalize, cast to f16.
// Assumes D == 256 (D/32 == 8 elems/lane; loops and v8h store fixed at 8).
__global__ void dml_normalize_rows(const float* __restrict__ desc,
                                   _Float16* __restrict__ nd,
                                   int rows, int D) {
    const int wave = (int)((blockIdx.x * blockDim.x + threadIdx.x) >> 5);
    const int lane = (int)(threadIdx.x & 31);
    if (wave >= rows) return;
    const int per = D >> 5;                       // elements per lane (8)
    const float* src = desc + (size_t)wave * D + (size_t)lane * per;

    float vals[8];
    float s = 0.0f;
#pragma unroll
    for (int i = 0; i < 8; ++i) {
        float v = src[i];
        vals[i] = v;
        s += v * v;
    }
#pragma unroll
    for (int off = 16; off > 0; off >>= 1) s += __shfl_xor(s, off, 32);

    const float nrm = sqrtf(s);
    const float scale = (nrm > 0.0f) ? (1.0f / nrm) : 0.0f;

    v8h out;
#pragma unroll
    for (int i = 0; i < 8; ++i) out[i] = (_Float16)(vals[i] * scale);
    *(v8h*)(nd + (size_t)wave * D + (size_t)lane * per) = out;
}

// One wave per job = (frame pair (a,b), row-tile tn, group of 4 col-tiles tg).
// Only jobs with 4*tg+3 >= tn exist (strict-upper-triangle pruning). All 4
// tiles of a group are computed unconditionally (straight-line WMMA); the
// n<m mask in the epilogue kills contributions from below-diagonal tiles.
template <int DFIX>
__global__ void dml_match_loss_wmma(const _Float16* __restrict__ nd,
                                    const float* __restrict__ pts_src,
                                    const float* __restrict__ pts_dst,
                                    const int* __restrict__ hptr,
                                    const int* __restrict__ wptr,
                                    float* __restrict__ acc,
                                    int Bn, int N, int Drt,
                                    int TG, int jobsPerPair, int totalJobs) {
    const int D = DFIX ? DFIX : Drt;

    // gwave is uniform per wave; pin it to an SGPR so the bounds check is a
    // scalar branch (EXEC never partially masked around WMMA).
    const int gwave = __builtin_amdgcn_readfirstlane(
        (int)((blockIdx.x * blockDim.x + threadIdx.x) >> 5));
    const int lane = (int)(threadIdx.x & 31);
    if (gwave >= totalJobs) return;

    // ---- decode job (all scalar) ----
    const int pair = gwave / jobsPerPair;
    int t          = gwave - pair * jobsPerPair;
    const int a = __builtin_amdgcn_readfirstlane(pair / Bn);
    const int b = __builtin_amdgcn_readfirstlane(pair - a * Bn);
    int tn_ = 0;
    while (t >= TG - (tn_ >> 2)) { t -= TG - (tn_ >> 2); ++tn_; }
    const int tn = __builtin_amdgcn_readfirstlane(tn_);
    const int tg = __builtin_amdgcn_readfirstlane((tn >> 2) + t);

    const int row_base = tn << 4;
    const int col_base = (tg << 2) << 4;     // first of 4 col tiles
    const int g   = lane >> 4;               // half-wave group
    const int c16 = lane & 15;

    const float sx = ((float)(wptr[0] - 1)) * 0.5f;   // denormalize scale (x)
    const float sy = ((float)(hptr[0] - 1)) * 0.5f;   // denormalize scale (y)

    // A fragment source: row = row_base + (lane&15), frame b (src descriptors)
    const _Float16* Arow = nd + ((size_t)b * N + row_base + c16) * D;
    // B fragment sources: col = col_base + 16*i + (lane&15), frame a
    const _Float16* Bcol = nd + ((size_t)a * N + col_base + c16) * D + 16 * g;

    v8f c0 = {}, c1 = {}, c2 = {}, c3 = {};

    // ---- K loop: fully unrolled when DFIX != 0 ----
#pragma unroll
    for (int kk = 0; kk < D; kk += 32) {
        // 16-bit A layout (16x32): lane holds K = kk+16g+0..15
        v8h alo = *(const v8h*)(Arow + kk + 16 * g);
        v8h ahi = *(const v8h*)(Arow + kk + 16 * g + 8);
        v16h Af = __builtin_shufflevector(alo, ahi,
            0,1,2,3,4,5,6,7,8,9,10,11,12,13,14,15);

        // 16-bit B layout (32x16): lane holds col n, K = kk+16g+0..15
        v8h b0l = *(const v8h*)(Bcol + (size_t)0 * 16 * D + kk);
        v8h b0h = *(const v8h*)(Bcol + (size_t)0 * 16 * D + kk + 8);
        v8h b1l = *(const v8h*)(Bcol + (size_t)1 * 16 * D + kk);
        v8h b1h = *(const v8h*)(Bcol + (size_t)1 * 16 * D + kk + 8);
        v8h b2l = *(const v8h*)(Bcol + (size_t)2 * 16 * D + kk);
        v8h b2h = *(const v8h*)(Bcol + (size_t)2 * 16 * D + kk + 8);
        v8h b3l = *(const v8h*)(Bcol + (size_t)3 * 16 * D + kk);
        v8h b3h = *(const v8h*)(Bcol + (size_t)3 * 16 * D + kk + 8);

        v16h B0 = __builtin_shufflevector(b0l, b0h, 0,1,2,3,4,5,6,7,8,9,10,11,12,13,14,15);
        v16h B1 = __builtin_shufflevector(b1l, b1h, 0,1,2,3,4,5,6,7,8,9,10,11,12,13,14,15);
        v16h B2 = __builtin_shufflevector(b2l, b2h, 0,1,2,3,4,5,6,7,8,9,10,11,12,13,14,15);
        v16h B3 = __builtin_shufflevector(b3l, b3h, 0,1,2,3,4,5,6,7,8,9,10,11,12,13,14,15);

        c0 = __builtin_amdgcn_wmma_f32_16x16x32_f16(false, Af, false, B0, (short)0, c0, false, false);
        c1 = __builtin_amdgcn_wmma_f32_16x16x32_f16(false, Af, false, B1, (short)0, c1, false, false);
        c2 = __builtin_amdgcn_wmma_f32_16x16x32_f16(false, Af, false, B2, (short)0, c2, false, false);
        c3 = __builtin_amdgcn_wmma_f32_16x16x32_f16(false, Af, false, B3, (short)0, c3, false, false);
    }

    // ---- fused epilogue: pixel-distance + triangle mask, local reduce ----
    // C layout: element v of v8f -> row M = v + 8g, col N = lane&15
    float psx[8], psy[8];
    const float* psrow = pts_src + ((size_t)b * N + row_base + 8 * g) * 2;
#pragma unroll
    for (int v = 0; v < 8; ++v) {
        psx[v] = (psrow[2 * v + 0] + 1.0f) * sx;
        psy[v] = (psrow[2 * v + 1] + 1.0f) * sy;
    }

    float lsum = 0.0f, lcnt = 0.0f;
    const float* pdbase = pts_dst + (((size_t)a * Bn + b) * N + col_base + c16) * 2;
#pragma unroll
    for (int i = 0; i < 4; ++i) {
        const int m = col_base + 16 * i + c16;
        const float pdx = (pdbase[32 * i + 0] + 1.0f) * sx;
        const float pdy = (pdbase[32 * i + 1] + 1.0f) * sy;
        const v8f cc = (i == 0) ? c0 : (i == 1) ? c1 : (i == 2) ? c2 : c3;
#pragma unroll
        for (int v = 0; v < 8; ++v) {
            const int n = row_base + 8 * g + v;
            const float dx = psx[v] - pdx;
            const float dy = psy[v] - pdy;
            const bool ok = (dx * dx + dy * dy <= 1.0f) & (n < m);
            lsum += ok ? (1.0f - cc[v]) : 0.0f;
            lcnt += ok ? 1.0f : 0.0f;
        }
    }

#pragma unroll
    for (int off = 16; off > 0; off >>= 1) {
        lsum += __shfl_xor(lsum, off, 32);
        lcnt += __shfl_xor(lcnt, off, 32);
    }
    if (lane == 0 && lcnt != 0.0f) {
        atomicAdd(&acc[0], lsum);
        atomicAdd(&acc[1], lcnt);
    }
}

__global__ void dml_finalize(const float* __restrict__ acc,
                             float* __restrict__ out) {
    if (threadIdx.x == 0 && blockIdx.x == 0) out[0] = acc[0] / acc[1];
}

extern "C" void kernel_launch(void* const* d_in, const int* in_sizes, int n_in,
                              void* d_out, int out_size, void* d_ws, size_t ws_size,
                              hipStream_t stream) {
    const float* desc    = (const float*)d_in[0];  // [B,N,D] f32
    const float* pts_src = (const float*)d_in[1];  // [B,N,2] f32
    const float* pts_dst = (const float*)d_in[2];  // [B,B,N,2] f32
    const int*   hptr    = (const int*)d_in[4];    // height (1 elem)
    const int*   wptr    = (const int*)d_in[5];    // width  (1 elem)
    float*       out     = (float*)d_out;

    // Derive shapes from flat sizes: in_sizes[2]=B*B*N*2, in_sizes[1]=B*N*2
    const int Bn = in_sizes[2] / in_sizes[1];
    const int N  = in_sizes[1] / (2 * Bn);
    const int D  = in_sizes[0] / (Bn * N);

    float*    acc = (float*)d_ws;
    _Float16* nd  = (_Float16*)((char*)d_ws + WS_HDR);

    // 1) zero accumulators (ws is poisoned / stale between calls)
    dml_init_accum<<<1, 32, 0, stream>>>(acc);

    // 2) normalize descriptor rows -> unit f16
    const int rows = Bn * N;
    dml_normalize_rows<<<(rows + 7) / 8, 256, 0, stream>>>(desc, nd, rows, D);

    // 3) fused WMMA GEMM + masked reduction over upper-triangle tile jobs
    const int TN = N / 16;            // 16-row tiles per dim
    const int TG = TN / 4;            // groups of 4 column tiles
    const int jobsPerPair = TN * TG - 2 * TG * (TG - 1);
    const int totalJobs   = Bn * Bn * jobsPerPair;
    const int blocks      = (totalJobs + 7) / 8;   // 8 waves / block (wave32)
    if (D == 256) {
        dml_match_loss_wmma<256><<<blocks, 256, 0, stream>>>(
            nd, pts_src, pts_dst, hptr, wptr, acc,
            Bn, N, D, TG, jobsPerPair, totalJobs);
    } else {
        dml_match_loss_wmma<0><<<blocks, 256, 0, stream>>>(
            nd, pts_src, pts_dst, hptr, wptr, acc,
            Bn, N, D, TG, jobsPerPair, totalJobs);
    }

    // 4) loss = sum / count
    dml_finalize<<<1, 32, 0, stream>>>(acc, out);
}
